// MultiHeadAttention_23390391894302
// MI455X (gfx1250) — matrix operand order repaired and checked by reading, not compiled, not run on hardware
//
#include <hip/hip_runtime.h>
#include <hip/hip_bf16.h>

// ---------------------------------------------------------------------------
// MI455X (gfx1250) fused MHA forward.
//   out  [B,S,D]  f32   (first 4,194,304 floats of d_out)
//   heat [H*B,S,S] f32  (next 134,217,728 floats of d_out)
// Bandwidth-bound on the 537MB heatmap write -> single pass over scores with
// online softmax; all matmuls via v_wmma_f32_16x16x32_f16 (f32 accumulate).
// Workspace: 25,165,824 halves = 48 MB.
// ---------------------------------------------------------------------------

typedef _Float16 h16;
typedef __attribute__((ext_vector_type(4)))  _Float16 v4h;
typedef __attribute__((ext_vector_type(8)))  _Float16 v8h;
typedef __attribute__((ext_vector_type(16))) _Float16 v16h;
typedef __attribute__((ext_vector_type(8)))  float    v8f;

#define WMMA16(A,B,C) __builtin_amdgcn_wmma_f32_16x16x32_f16(false,(A),false,(B),(short)0,(C),false,false)

static constexpr int H_ = 16, B_ = 2, S_ = 2048, D_ = 1024;

__device__ __forceinline__ v8f v8f_zero() {
  v8f z;
#pragma unroll
  for (int i = 0; i < 8; ++i) z[i] = 0.0f;
  return z;
}

// A-fragment (16 rows x 32 k, f16) from a row-major [M,K] LDS tile.
// ISA 7.12.2 16-bit A layout: lane<16 holds row=lane, K {k0..k0+7, k0+16..k0+23};
// lane>=16 holds row=lane-16, K {k0+8..k0+15, k0+24..k0+31}.
__device__ __forceinline__ v16h frag_a(const h16* p, int ld) {
  const int lane = threadIdx.x & 31;
  const int hi = lane >> 4, r = lane & 15;
  const h16* q = p + r * ld + hi * 8;
  v8h lo = *(const v8h*)q;
  v8h hh = *(const v8h*)(q + 16);
  return __builtin_shufflevector(lo, hh, 0,1,2,3,4,5,6,7,8,9,10,11,12,13,14,15);
}

// B-fragment (32 k x 16 cols, f16) from an [N,K]-stored (transposed) LDS tile.
// lane<16 holds col=lane, K {k0..k0+7, k0+16..k0+23}; lane>=16 col=lane-16, K {k0+8..k0+15, k0+24..k0+31}.
__device__ __forceinline__ v16h frag_b(const h16* p, int ld) {
  const int lane = threadIdx.x & 31;
  const int kh = lane >> 4, c = lane & 15;
  const h16* q = p + c * ld + kh * 8;
  v8h lo = *(const v8h*)q;
  v8h hh = *(const v8h*)(q + 16);
  return __builtin_shufflevector(lo, hh, 0,1,2,3,4,5,6,7,8,9,10,11,12,13,14,15);
}

// ---------------------------------------------------------------------------
// f32 -> f16 conversion, 4 elements/thread
// ---------------------------------------------------------------------------
__global__ void cvt_f32_to_f16(const float* __restrict__ s, h16* __restrict__ d, int n4) {
  int i = blockIdx.x * blockDim.x + threadIdx.x;
  if (i < n4) {
    float4 f = ((const float4*)s)[i];
    v4h o;
    o[0] = (h16)f.x; o[1] = (h16)f.y; o[2] = (h16)f.z; o[3] = (h16)f.w;
    ((v4h*)d)[i] = o;
  }
}

// ---------------------------------------------------------------------------
// Shared GEMM macro-tile: C[128x64] += A[128xK] * W[KxN], K=N stride=1024.
// 256 threads = 8 waves, waves as 4(M) x 2(N), each wave 32x32 = 2x2 WMMA tiles.
// ---------------------------------------------------------------------------
__device__ __forceinline__ void gemm_mac_128x64(
    const h16* __restrict__ A, const h16* __restrict__ Wt,
    int m0, int n0, h16* As, h16* Bs, v8f acc[2][2]) {
  const int tid = threadIdx.x;
  const int wid = tid >> 5;
  const int wm = wid & 3, wn = wid >> 2;
#pragma unroll
  for (int i = 0; i < 2; ++i)
#pragma unroll
    for (int j = 0; j < 2; ++j) acc[i][j] = v8f_zero();

  for (int kt = 0; kt < 32; ++kt) {
    if (kt + 1 < 32)  // global_prefetch_b8 of next A k-slice
      __builtin_prefetch(&A[(size_t)(m0 + (tid >> 1)) * 1024 + (kt + 1) * 32], 0, 1);
    __syncthreads();
    // A tile 128x32 -> As[row*40 + k] (pad 40 halves: 16B-aligned, bank-skewed)
#pragma unroll
    for (int i = tid; i < 512; i += 256) {
      int r = i >> 2, c = (i & 3) * 8;
      *(v8h*)&As[r * 40 + c] = *(const v8h*)&A[(size_t)(m0 + r) * 1024 + kt * 32 + c];
    }
    // W tile 32x64, stored transposed Bs[n*40 + k]
    {
      int k = tid >> 3, nc = (tid & 7) * 8;
      v8h w = *(const v8h*)&Wt[(size_t)(kt * 32 + k) * 1024 + n0 + nc];
#pragma unroll
      for (int e = 0; e < 8; ++e) Bs[(nc + e) * 40 + k] = w[e];
    }
    __syncthreads();
    v16h a0 = frag_a(&As[(wm * 32) * 40], 40);
    v16h a1 = frag_a(&As[(wm * 32 + 16) * 40], 40);
    v16h b0 = frag_b(&Bs[(wn * 32) * 40], 40);
    v16h b1 = frag_b(&Bs[(wn * 32 + 16) * 40], 40);
    acc[0][0] = WMMA16(a0, b0, acc[0][0]);
    acc[0][1] = WMMA16(a0, b1, acc[0][1]);
    acc[1][0] = WMMA16(a1, b0, acc[1][0]);
    acc[1][1] = WMMA16(a1, b1, acc[1][1]);
  }
}

// ---------------------------------------------------------------------------
// Q/K/V projection: [4096,1024] x [1024,1024] + bias -> f16 head-major [h,b,s,64]
// blockIdx.z selects Q / K / V.
// ---------------------------------------------------------------------------
__launch_bounds__(256)
__global__ void qkv_gemm_kernel(
    const h16* __restrict__ xh,
    const h16* __restrict__ Wqh, const h16* __restrict__ Wkh, const h16* __restrict__ Wvh,
    const float* __restrict__ bq, const float* __restrict__ bk, const float* __restrict__ bv,
    h16* __restrict__ Qd, h16* __restrict__ Kd, h16* __restrict__ Vd) {
  __shared__ __align__(16) h16 As[128 * 40];
  __shared__ __align__(16) h16 Bs[64 * 40];
  const h16* W; const float* bias; h16* dst;
  if (blockIdx.z == 0)      { W = Wqh; bias = bq; dst = Qd; }
  else if (blockIdx.z == 1) { W = Wkh; bias = bk; dst = Kd; }
  else                      { W = Wvh; bias = bv; dst = Vd; }

  const int m0 = blockIdx.x * 128, n0 = blockIdx.y * 64;
  v8f acc[2][2];
  gemm_mac_128x64(xh, W, m0, n0, As, Bs, acc);

  const int tid = threadIdx.x, wid = tid >> 5;
  const int wm = wid & 3, wn = wid >> 2;
  const int lane = tid & 31, hi = lane >> 4, cl = lane & 15;
#pragma unroll
  for (int tm = 0; tm < 2; ++tm)
#pragma unroll
    for (int tn = 0; tn < 2; ++tn)
#pragma unroll
      for (int r = 0; r < 8; ++r) {
        int row = m0 + wm * 32 + tm * 16 + hi * 8 + r;   // 0..4095  (= b*S + s)
        int col = n0 + wn * 32 + tn * 16 + cl;           // 0..1023  (= h*64 + d)
        float v = acc[tm][tn][r] + bias[col];
        int head = col >> 6, c64 = col & 63, bb = row >> 11, ss = row & 2047;
        dst[(((size_t)(head * B_ + bb)) * S_ + ss) * 64 + c64] = (h16)v;
      }
}

// ---------------------------------------------------------------------------
// Output projection: wV[4096,1024] x Wo[1024,1024] + bo -> f32 out [B,S,D]
// ---------------------------------------------------------------------------
__launch_bounds__(256)
__global__ void out_gemm_kernel(const h16* __restrict__ wVh, const h16* __restrict__ Woh,
                                const float* __restrict__ bo, float* __restrict__ out) {
  __shared__ __align__(16) h16 As[128 * 40];
  __shared__ __align__(16) h16 Bs[64 * 40];
  const int m0 = blockIdx.x * 128, n0 = blockIdx.y * 64;
  v8f acc[2][2];
  gemm_mac_128x64(wVh, Woh, m0, n0, As, Bs, acc);

  const int tid = threadIdx.x, wid = tid >> 5;
  const int wm = wid & 3, wn = wid >> 2;
  const int lane = tid & 31, hi = lane >> 4, cl = lane & 15;
#pragma unroll
  for (int tm = 0; tm < 2; ++tm)
#pragma unroll
    for (int tn = 0; tn < 2; ++tn)
#pragma unroll
      for (int r = 0; r < 8; ++r) {
        int row = m0 + wm * 32 + tm * 16 + hi * 8 + r;
        int col = n0 + wn * 32 + tn * 16 + cl;
        out[(size_t)row * 1024 + col] = acc[tm][tn][r] + bo[col];
      }
}

// ---------------------------------------------------------------------------
// Fused causal attention with online softmax + raw-score heatmap emission.
// grid: (H*B, S/64); block: 128 threads = 4 waves, 16 rows/wave.
// ---------------------------------------------------------------------------
__launch_bounds__(128)
__global__ void attn_kernel(const h16* __restrict__ Qh, const h16* __restrict__ Kh,
                            const h16* __restrict__ Vh, float* __restrict__ heat,
                            h16* __restrict__ wVh) {
  const int hb = blockIdx.x;            // h*B + b
  const int hd = hb >> 1, b = hb & 1;
  const int row0 = blockIdx.y * 64;
  const int tid = threadIdx.x, wid = tid >> 5, lane = tid & 31;
  const int hi = lane >> 4, cl = lane & 15;

  __shared__ __align__(16) h16 Qs[64 * 72];
  __shared__ __align__(16) h16 Ks[64 * 72];       // [col][dq]  (== B^T for Q*K^T)
  __shared__ __align__(16) h16 Vs[64 * 72];       // [vcol][t]  (== B^T for P*V)
  __shared__ __align__(16) h16 Ps[4 * 16 * 72];   // per-wave P rows

  const h16* Qg = Qh + (size_t)hb * S_ * 64;
  const h16* Kg = Kh + (size_t)hb * S_ * 64;
  const h16* Vg = Vh + (size_t)hb * S_ * 64;
  float*     hm = heat + (size_t)hb * S_ * S_;

  // Q row-block -> LDS
  for (int i = tid; i < 64 * 8; i += 128) {
    int r = i >> 3, c = (i & 7) * 8;
    *(v8h*)&Qs[r * 72 + c] = *(const v8h*)&Qg[(size_t)(row0 + r) * 64 + c];
  }

  float mrow[8], lrow[8];
  v8f O[4];
#pragma unroll
  for (int r = 0; r < 8; ++r) { mrow[r] = -1e30f; lrow[r] = 0.0f; }
#pragma unroll
  for (int t = 0; t < 4; ++t) O[t] = v8f_zero();

  for (int jb = 0; jb < 32; ++jb) {
    const int j0 = jb * 64;
    __syncthreads();
    for (int i = tid; i < 64 * 8; i += 128) {           // K block (row-major)
      int r = i >> 3, c = (i & 7) * 8;
      *(v8h*)&Ks[r * 72 + c] = *(const v8h*)&Kg[(size_t)(j0 + r) * 64 + c];
    }
    for (int i = tid; i < 64 * 64; i += 128) {          // V block, transposed
      int t = i >> 6, vc = i & 63;
      Vs[vc * 72 + t] = Vg[(size_t)(j0 + t) * 64 + vc];
    }
    __syncthreads();

    // scores: 16 rows x 64 cols per wave, k = 64 in two WMMA steps
    v16h aq0 = frag_a(&Qs[(wid * 16) * 72 + 0], 72);
    v16h aq1 = frag_a(&Qs[(wid * 16) * 72 + 32], 72);
    v8f st[4];
#pragma unroll
    for (int t = 0; t < 4; ++t) {
      v8f s = v8f_zero();
      s = WMMA16(aq0, frag_b(&Ks[(t * 16) * 72 + 0], 72), s);
      s = WMMA16(aq1, frag_b(&Ks[(t * 16) * 72 + 32], 72), s);
      st[t] = s;
    }

    // raw (pre-mask) scores -> heatmap
#pragma unroll
    for (int t = 0; t < 4; ++t)
#pragma unroll
      for (int r = 0; r < 8; ++r) {
        int rg = row0 + wid * 16 + hi * 8 + r;
        int cg = j0 + t * 16 + cl;
        hm[(size_t)rg * S_ + cg] = st[t][r];
      }

    if (j0 <= row0) {                     // block intersects causal region
      // causal mask (in place)
#pragma unroll
      for (int t = 0; t < 4; ++t)
#pragma unroll
        for (int r = 0; r < 8; ++r) {
          int rg = row0 + wid * 16 + hi * 8 + r;
          int cg = j0 + t * 16 + cl;
          if (cg > rg) st[t][r] = -1e30f;
        }
      // row max across 64 cols (4 tiles, then across the 16 lanes of each half)
      float mx[8];
#pragma unroll
      for (int r = 0; r < 8; ++r)
        mx[r] = fmaxf(fmaxf(st[0][r], st[1][r]), fmaxf(st[2][r], st[3][r]));
#pragma unroll
      for (int off = 1; off < 16; off <<= 1)
#pragma unroll
        for (int r = 0; r < 8; ++r) mx[r] = fmaxf(mx[r], __shfl_xor(mx[r], off, 16));

      float sc[8], rs[8];
#pragma unroll
      for (int r = 0; r < 8; ++r) {
        float nm = fmaxf(mrow[r], mx[r]);
        sc[r] = __expf(mrow[r] - nm);
        mrow[r] = nm;
        rs[r] = 0.0f;
      }
#pragma unroll
      for (int t = 0; t < 4; ++t)
#pragma unroll
        for (int r = 0; r < 8; ++r) {
          float p = __expf(st[t][r] - mrow[r]);
          st[t][r] = p;
          rs[r] += p;
        }
#pragma unroll
      for (int off = 1; off < 16; off <<= 1)
#pragma unroll
        for (int r = 0; r < 8; ++r) rs[r] += __shfl_xor(rs[r], off, 16);
#pragma unroll
      for (int r = 0; r < 8; ++r) lrow[r] = lrow[r] * sc[r] + rs[r];
#pragma unroll
      for (int t = 0; t < 4; ++t)
#pragma unroll
        for (int r = 0; r < 8; ++r) O[t][r] *= sc[r];

      // P -> per-wave LDS (C-layout -> A-layout turnaround)
      h16* Pw = &Ps[(wid * 16) * 72];
#pragma unroll
      for (int t = 0; t < 4; ++t)
#pragma unroll
        for (int r = 0; r < 8; ++r)
          Pw[(hi * 8 + r) * 72 + t * 16 + cl] = (h16)st[t][r];
      asm volatile("s_wait_dscnt 0x0" ::: "memory");   // wave-local LDS RAW

      // O += P @ V   (k = 64 in two WMMA steps)
#pragma unroll
      for (int kk = 0; kk < 2; ++kk) {
        v16h ap = frag_a(&Pw[kk * 32], 72);
#pragma unroll
        for (int t2 = 0; t2 < 4; ++t2)
          O[t2] = WMMA16(ap, frag_b(&Vs[(t2 * 16) * 72 + kk * 32], 72), O[t2]);
      }
    }
  }

  // normalize and store wV as f16 row-major [b*S+s, h*64+v]
  float inv[8];
#pragma unroll
  for (int r = 0; r < 8; ++r) inv[r] = 1.0f / lrow[r];
#pragma unroll
  for (int t2 = 0; t2 < 4; ++t2)
#pragma unroll
    for (int r = 0; r < 8; ++r) {
      int rg = row0 + wid * 16 + hi * 8 + r;
      int cg = t2 * 16 + cl;
      wVh[(size_t)(b * S_ + rg) * 1024 + hd * 64 + cg] = (h16)(O[t2][r] * inv[r]);
    }
}

// ---------------------------------------------------------------------------
extern "C" void kernel_launch(void* const* d_in, const int* in_sizes, int n_in,
                              void* d_out, int out_size, void* d_ws, size_t ws_size,
                              hipStream_t stream) {
  const float* x  = (const float*)d_in[0];
  const float* Wq = (const float*)d_in[1];
  const float* bq = (const float*)d_in[2];
  const float* Wk = (const float*)d_in[3];
  const float* bk = (const float*)d_in[4];
  const float* Wv = (const float*)d_in[5];
  const float* bv = (const float*)d_in[6];
  const float* Wo = (const float*)d_in[7];
  const float* bo = (const float*)d_in[8];

  const size_t MSZ = (size_t)B_ * S_ * D_;   // 4,194,304
  const size_t WSZ = (size_t)D_ * D_;        // 1,048,576

  h16* p = (h16*)d_ws;
  h16* xh  = p; p += MSZ;
  h16* Wqh = p; p += WSZ;
  h16* Wkh = p; p += WSZ;
  h16* Wvh = p; p += WSZ;
  h16* Woh = p; p += WSZ;
  h16* Qh  = p; p += MSZ;
  h16* Kh  = p; p += MSZ;
  h16* Vh  = p; p += MSZ;
  h16* wVh = p; p += MSZ;

  float* out  = (float*)d_out;
  float* heat = out + MSZ;

  cvt_f32_to_f16<<<(int)(MSZ / 4 / 256), 256, 0, stream>>>(x,  xh,  (int)(MSZ / 4));
  cvt_f32_to_f16<<<(int)(WSZ / 4 / 256), 256, 0, stream>>>(Wq, Wqh, (int)(WSZ / 4));
  cvt_f32_to_f16<<<(int)(WSZ / 4 / 256), 256, 0, stream>>>(Wk, Wkh, (int)(WSZ / 4));
  cvt_f32_to_f16<<<(int)(WSZ / 4 / 256), 256, 0, stream>>>(Wv, Wvh, (int)(WSZ / 4));
  cvt_f32_to_f16<<<(int)(WSZ / 4 / 256), 256, 0, stream>>>(Wo, Woh, (int)(WSZ / 4));

  qkv_gemm_kernel<<<dim3(32, 16, 3), 256, 0, stream>>>(
      xh, Wqh, Wkh, Wvh, bq, bk, bv, Qh, Kh, Vh);

  attn_kernel<<<dim3(H_ * B_, S_ / 64), 128, 0, stream>>>(Qh, Kh, Vh, heat, wVh);

  out_gemm_kernel<<<dim3(32, 16), 256, 0, stream>>>(wVh, Woh, bo, out);
}
